// GATv2_20203526160489
// MI455X (gfx1250) — matrix operand order repaired and checked by reading, not compiled, not run on hardware
//
#include <hip/hip_runtime.h>
#include <math.h>

// ---------------------------------------------------------------------------
// GATv2 (3 layers + final linear) for gfx1250.
// Dense GEMMs: bf16 WMMA (v_wmma_f32_16x16x32_bf16), f32 accumulate,
//   register-blocked 32x64 per wave (2 M-tiles x 4 N-tiles, 8 WMMAs/k-step).
// Edge attention/softmax/scatter: f32 with global atomics (L2-resident).
// ---------------------------------------------------------------------------

typedef __attribute__((ext_vector_type(16))) __bf16          v16bf;
typedef __attribute__((ext_vector_type(8)))  float           v8f;
typedef __attribute__((ext_vector_type(16))) unsigned short  v16u;
typedef __attribute__((ext_vector_type(8)))  unsigned short  v8u;

#define NEG_SLOPE 0.2f

__device__ __forceinline__ unsigned short f32_to_bf16_rne(float f) {
  unsigned u = __float_as_uint(f);
  unsigned r = 0x7FFFu + ((u >> 16) & 1u);
  return (unsigned short)((u + r) >> 16);
}

__device__ __forceinline__ void atomicMaxF(float* addr, float v) {
  // classic sign-split trick; works with -inf (0xFF800000) initialization
  if (v >= 0.0f) atomicMax((int*)addr, __float_as_int(v));
  else           atomicMin((unsigned int*)addr, __float_as_uint(v));
}

// ------------------------------ conversions --------------------------------

__global__ void k_f32_to_bf16(const float* __restrict__ in,
                              unsigned short* __restrict__ out, long n) {
  long i = (long)blockIdx.x * blockDim.x + threadIdx.x;
  if (i < n) out[i] = f32_to_bf16_rne(in[i]);
}

// W: K x N row-major f32  ->  Wt: N x K row-major bf16 (transposed)
__global__ void k_w_to_bf16_t(const float* __restrict__ W,
                              unsigned short* __restrict__ Wt, int K, int N) {
  int i = blockIdx.x * blockDim.x + threadIdx.x;
  if (i >= K * N) return;
  int k = i / N, n = i - k * N;
  Wt[(size_t)n * K + k] = f32_to_bf16_rne(W[i]);
}

__global__ void k_fill(float* __restrict__ p, float v, long n) {
  long i = (long)blockIdx.x * blockDim.x + threadIdx.x;
  if (i < n) p[i] = v;
}

// ------------------------------ WMMA GEMM ----------------------------------
// C[M,N] = A[M,K] (bf16 row-major) * B[K,N], with B supplied transposed as
// Bt[N,K] bf16. One wave computes a 32x64 block of C: 2 M-tiles x 4 N-tiles,
// 8 accumulators, 6 fragment loads -> 8 WMMAs per 32-wide k-step.
// Requires: M % 32 == 0, N % 64 == 0, K % 32 == 0 (true for all layers).

__global__ void k_gemm_wmma_bf16(const unsigned short* __restrict__ A,
                                 const unsigned short* __restrict__ Bt,
                                 float* __restrict__ C, int M, int N, int K) {
  int wave = (int)((blockIdx.x * blockDim.x + threadIdx.x) >> 5);
  int lane = threadIdx.x & 31;
  int nbn  = N >> 6;                        // # 64-column blocks
  int mb   = wave / nbn;
  int nb   = wave - mb * nbn;
  if (mb >= (M >> 5)) return;               // wave-uniform exit (EXEC stays full)

  int m0 = mb << 5;                         // 32-row block base
  int n0 = nb << 6;                         // 64-col block base
  int l15 = lane & 15;
  int ka = (lane & 16) ? 16 : 0;            // A: lane halves hold K 0..15 / 16..31
  int kb = (lane & 16) ? 16 : 0;            // B: lane halves hold K 0..15 / 16..31

  const unsigned short* ap0 = A  + (size_t)(m0 + l15)      * K + ka;
  const unsigned short* ap1 = A  + (size_t)(m0 + 16 + l15) * K + ka;
  const unsigned short* bp0 = Bt + (size_t)(n0 + l15)      * K + kb;
  const unsigned short* bp1 = Bt + (size_t)(n0 + 16 + l15) * K + kb;
  const unsigned short* bp2 = Bt + (size_t)(n0 + 32 + l15) * K + kb;
  const unsigned short* bp3 = Bt + (size_t)(n0 + 48 + l15) * K + kb;

  v8f acc[2][4];
#pragma unroll
  for (int i = 0; i < 2; ++i)
#pragma unroll
    for (int j = 0; j < 4; ++j) acc[i][j] = (v8f){};

  for (int k0 = 0; k0 < K; k0 += 32) {
    v16bf a[2], b[4];
    {
      v8u lo0 = *(const v8u*)(ap0 + k0), hi0 = *(const v8u*)(ap0 + k0 + 8);
      v8u lo1 = *(const v8u*)(ap1 + k0), hi1 = *(const v8u*)(ap1 + k0 + 8);
      v16u a0, a1;
#pragma unroll
      for (int i = 0; i < 8; ++i) {
        a0[i] = lo0[i]; a0[i + 8] = hi0[i];
        a1[i] = lo1[i]; a1[i + 8] = hi1[i];
      }
      a[0] = __builtin_bit_cast(v16bf, a0);
      a[1] = __builtin_bit_cast(v16bf, a1);
      b[0] = __builtin_bit_cast(v16bf, *(const v16u*)(bp0 + k0));
      b[1] = __builtin_bit_cast(v16bf, *(const v16u*)(bp1 + k0));
      b[2] = __builtin_bit_cast(v16bf, *(const v16u*)(bp2 + k0));
      b[3] = __builtin_bit_cast(v16bf, *(const v16u*)(bp3 + k0));
    }
#pragma unroll
    for (int i = 0; i < 2; ++i)
#pragma unroll
      for (int j = 0; j < 4; ++j)
        acc[i][j] = __builtin_amdgcn_wmma_f32_16x16x32_bf16(
            false, a[i], false, b[j], (short)0, acc[i][j], false, false);
  }

  // C/D layout: VGPR r -> row tile_m0 + r + 8*(lane>=16), col tile_n0 + lane%16
  int rbase = ((lane & 16) ? 8 : 0);
#pragma unroll
  for (int i = 0; i < 2; ++i) {
#pragma unroll
    for (int j = 0; j < 4; ++j) {
      float* cp = C + (size_t)(m0 + i * 16 + rbase) * N + (n0 + j * 16 + l15);
#pragma unroll
      for (int r = 0; r < 8; ++r) cp[(size_t)r * N] = acc[i][j][r];
    }
  }
}

// ------------------------------ edge phase ---------------------------------
// logits[e,h] = att_h . leaky_relu(xl[src,h,:] + xr[dst,h,:]); fused seg-max.
__global__ void k_edge_logits(const float* __restrict__ xl,
                              const float* __restrict__ xr,
                              const float* __restrict__ att,
                              const int* __restrict__ srcE,
                              const int* __restrict__ dstE,
                              int E, int Etot, int H, int C,
                              float* __restrict__ logits,
                              float* __restrict__ mbuf) {
  int idx = blockIdx.x * blockDim.x + threadIdx.x;
  if (idx >= Etot * H) return;
  int h = idx % H, e = idx / H;
  int s = (e < E) ? srcE[e] : (e - E);      // self-loops appended
  int d = (e < E) ? dstE[e] : (e - E);
  const float4* pl = (const float4*)(xl + (size_t)s * H * C + (size_t)h * C);
  const float4* pr = (const float4*)(xr + (size_t)d * H * C + (size_t)h * C);
  const float4* pa = (const float4*)(att + (size_t)h * C);
  float acc = 0.0f;
  for (int c = 0; c < (C >> 2); ++c) {
    float4 l = pl[c], r = pr[c], a = pa[c];
    float v0 = l.x + r.x, v1 = l.y + r.y, v2 = l.z + r.z, v3 = l.w + r.w;
    v0 = (v0 > 0.0f) ? v0 : NEG_SLOPE * v0;
    v1 = (v1 > 0.0f) ? v1 : NEG_SLOPE * v1;
    v2 = (v2 > 0.0f) ? v2 : NEG_SLOPE * v2;
    v3 = (v3 > 0.0f) ? v3 : NEG_SLOPE * v3;
    acc = fmaf(a.x, v0, acc); acc = fmaf(a.y, v1, acc);
    acc = fmaf(a.z, v2, acc); acc = fmaf(a.w, v3, acc);
  }
  logits[idx] = acc;
  atomicMaxF(mbuf + (size_t)d * H + h, acc);
}

__global__ void k_edge_exp(const float* __restrict__ logits,
                           const float* __restrict__ mbuf,
                           float* __restrict__ den,
                           const int* __restrict__ dstE,
                           int E, int Etot, int H,
                           float* __restrict__ aexp) {
  int idx = blockIdx.x * blockDim.x + threadIdx.x;
  if (idx >= Etot * H) return;
  int h = idx % H, e = idx / H;
  int d = (e < E) ? dstE[e] : (e - E);
  float a = expf(logits[idx] - mbuf[(size_t)d * H + h]);
  aexp[idx] = a;
  atomicAdd(den + (size_t)d * H + h, a);
}

// out[dst,h,c0..c0+3] += (aexp/den[dst]) * xl[src,h,c0..c0+3]   (C % 4 == 0)
__global__ void k_edge_scatter(const float* __restrict__ xl,
                               const float* __restrict__ aexp,
                               const float* __restrict__ den,
                               const int* __restrict__ srcE,
                               const int* __restrict__ dstE,
                               int E, int Etot, int H, int C,
                               float* __restrict__ out) {
  long idx = (long)blockIdx.x * blockDim.x + threadIdx.x;
  int C4 = C >> 2;
  long total = (long)Etot * H * C4;
  if (idx >= total) return;
  int c4 = (int)(idx % C4);
  long eh = idx / C4;
  int h = (int)(eh % H);
  int e = (int)(eh / H);
  int s = (e < E) ? srcE[e] : (e - E);
  int d = (e < E) ? dstE[e] : (e - E);
  float alpha = aexp[(size_t)e * H + h] / den[(size_t)d * H + h];
  float4 v = *(const float4*)(xl + (size_t)s * H * C + (size_t)h * C + 4 * c4);
  float* op = out + (size_t)d * H * C + (size_t)h * C + 4 * c4;
  atomicAdd(op + 0, alpha * v.x);
  atomicAdd(op + 1, alpha * v.y);
  atomicAdd(op + 2, alpha * v.z);
  atomicAdd(op + 3, alpha * v.w);
}

__global__ void k_bias_relu(const float* __restrict__ in,
                            const float* __restrict__ b,
                            float* __restrict__ out, int Nn, int F) {
  long i = (long)blockIdx.x * blockDim.x + threadIdx.x;
  if (i >= (long)Nn * F) return;
  int j = (int)(i % F);
  float v = in[i] + b[j];
  out[i] = (v > 0.0f) ? v : 0.0f;
}

__global__ void k_final(const float* __restrict__ h,
                        const float* __restrict__ Wf,
                        const float* __restrict__ bf_,
                        float* __restrict__ out, int Nn, int K) {
  int n = blockIdx.x * blockDim.x + threadIdx.x;
  if (n >= Nn) return;
  const float4* p = (const float4*)(h + (size_t)n * K);
  const float4* w = (const float4*)Wf;
  float acc = 0.0f;
  for (int k = 0; k < (K >> 2); ++k) {
    float4 a = p[k], b = w[k];
    acc = fmaf(a.x, b.x, acc); acc = fmaf(a.y, b.y, acc);
    acc = fmaf(a.z, b.z, acc); acc = fmaf(a.w, b.w, acc);
  }
  out[n] = acc + bf_[0];
}

// ------------------------------ host driver --------------------------------

static inline int cdiv(long a, int b) { return (int)((a + b - 1) / b); }

static void run_layer(const float* in, int Fin, int H, int C,
                      const float* Wl, const float* Wr,
                      const float* att, const float* bias,
                      const int* srcE, const int* dstE, int E, int Nn,
                      unsigned short* abf, unsigned short* wlb, unsigned short* wrb,
                      float* xl, float* xr, float* outb,
                      float* logits, float* aexp, float* mb, float* den,
                      float* hout, hipStream_t stream) {
  int HC = H * C;
  int Etot = E + Nn;

  long na = (long)Nn * Fin;
  k_f32_to_bf16<<<cdiv(na, 256), 256, 0, stream>>>(in, abf, na);
  k_w_to_bf16_t<<<cdiv((long)Fin * HC, 256), 256, 0, stream>>>(Wl, wlb, Fin, HC);
  k_w_to_bf16_t<<<cdiv((long)Fin * HC, 256), 256, 0, stream>>>(Wr, wrb, Fin, HC);

  int waves  = (Nn / 32) * (HC / 64);           // Nn = 20000 = 625*32; HC%64==0
  int blocks = cdiv((long)waves * 32, 256);
  k_gemm_wmma_bf16<<<blocks, 256, 0, stream>>>(abf, wlb, xl, Nn, HC, Fin);
  k_gemm_wmma_bf16<<<blocks, 256, 0, stream>>>(abf, wrb, xr, Nn, HC, Fin);

  long nmh = (long)Nn * H;
  k_fill<<<cdiv(nmh, 256), 256, 0, stream>>>(mb, -__builtin_inff(), nmh);
  k_fill<<<cdiv(nmh, 256), 256, 0, stream>>>(den, 0.0f, nmh);
  long nout = (long)Nn * HC;
  k_fill<<<cdiv(nout, 256), 256, 0, stream>>>(outb, 0.0f, nout);

  int ehn = Etot * H;
  k_edge_logits<<<cdiv(ehn, 256), 256, 0, stream>>>(xl, xr, att, srcE, dstE,
                                                    E, Etot, H, C, logits, mb);
  k_edge_exp<<<cdiv(ehn, 256), 256, 0, stream>>>(logits, mb, den, dstE,
                                                 E, Etot, H, aexp);
  long tot4 = (long)Etot * H * (C >> 2);
  k_edge_scatter<<<cdiv(tot4, 256), 256, 0, stream>>>(xl, aexp, den, srcE, dstE,
                                                      E, Etot, H, C, outb);
  k_bias_relu<<<cdiv(nout, 256), 256, 0, stream>>>(outb, bias, hout, Nn, HC);
}

extern "C" void kernel_launch(void* const* d_in, const int* in_sizes, int n_in,
                              void* d_out, int out_size, void* d_ws, size_t ws_size,
                              hipStream_t stream) {
  const float* x    = (const float*)d_in[0];
  const int*   ei   = (const int*)  d_in[1];
  const float* Wl1  = (const float*)d_in[2];
  const float* Wr1  = (const float*)d_in[3];
  const float* att1 = (const float*)d_in[4];
  const float* b1   = (const float*)d_in[5];
  const float* Wl2  = (const float*)d_in[6];
  const float* Wr2  = (const float*)d_in[7];
  const float* att2 = (const float*)d_in[8];
  const float* b2   = (const float*)d_in[9];
  const float* Wl3  = (const float*)d_in[10];
  const float* Wr3  = (const float*)d_in[11];
  const float* att3 = (const float*)d_in[12];
  const float* b3   = (const float*)d_in[13];
  const float* Wf   = (const float*)d_in[14];
  const float* bf_  = (const float*)d_in[15];

  const int Nn = in_sizes[0] / 64;   // 20000
  const int E  = in_sizes[1] / 2;    // 320000
  const int* srcE = ei;
  const int* dstE = ei + E;
  const int Etot = E + Nn;
  const int FMAX = 768;

  char* ws = (char*)d_ws;
  size_t off = 0;
  auto alloc = [&](size_t bytes) -> void* {
    void* p = ws + off;
    off = (off + bytes + 255) & ~(size_t)255;
    return p;
  };
  float*          hbuf   = (float*)alloc((size_t)Nn * FMAX * 4);
  float*          xl     = (float*)alloc((size_t)Nn * FMAX * 4);
  float*          xr     = (float*)alloc((size_t)Nn * FMAX * 4);
  float*          outb   = (float*)alloc((size_t)Nn * FMAX * 4);
  unsigned short* abf    = (unsigned short*)alloc((size_t)Nn * FMAX * 2);
  unsigned short* wlb    = (unsigned short*)alloc((size_t)FMAX * FMAX * 2);
  unsigned short* wrb    = (unsigned short*)alloc((size_t)FMAX * FMAX * 2);
  float*          logits = (float*)alloc((size_t)Etot * 3 * 4);
  float*          aexp   = (float*)alloc((size_t)Etot * 3 * 4);
  float*          mb     = (float*)alloc((size_t)Nn * 3 * 4);
  float*          den    = (float*)alloc((size_t)Nn * 3 * 4);

  // Layer 1: 64 -> 3 heads x 64 (concat 192)
  run_layer(x, 64, 3, 64, Wl1, Wr1, att1, b1, srcE, dstE, E, Nn,
            abf, wlb, wrb, xl, xr, outb, logits, aexp, mb, den, hbuf, stream);
  // Layer 2: 192 -> 3 heads x 256 (concat 768)
  run_layer(hbuf, 192, 3, 256, Wl2, Wr2, att2, b2, srcE, dstE, E, Nn,
            abf, wlb, wrb, xl, xr, outb, logits, aexp, mb, den, hbuf, stream);
  // Layer 3: 768 -> 1 head x 512
  run_layer(hbuf, 768, 1, 512, Wl3, Wr3, att3, b3, srcE, dstE, E, Nn,
            abf, wlb, wrb, xl, xr, outb, logits, aexp, mb, den, hbuf, stream);
  // Final linear 512 -> 1
  k_final<<<cdiv(Nn, 256), 256, 0, stream>>>(hbuf, Wf, bf_, (float*)d_out, Nn, 512);
}
